// MambaBlock_37366215475565
// MI455X (gfx1250) — matrix operand order repaired and checked by reading, not compiled, not run on hardware
//
#include <hip/hip_runtime.h>
#include <hip/hip_bf16.h>

// ---------------- problem constants ----------------
#define B_   8
#define L_   2048
#define D_   1024
#define DI   2048
#define Nst  16
#define DTR  64
#define Kcv  4
#define BT   (B_ * L_)          // 16384 tokens
#define EPSf 1e-5f
#define NCH  32                 // scan chunks
#define LC   (L_ / NCH)         // 64 timesteps per chunk

// GEMM register tiling: each wave owns a (16*MT) x (16*NT) output tile.
#define MT   4
#define NT   2

typedef __attribute__((ext_vector_type(16))) _Float16 v16h;
typedef __attribute__((ext_vector_type(8)))  _Float16 v8h;
typedef __attribute__((ext_vector_type(8)))  float    v8f;

// ---------------- fp32 -> fp16 cast ----------------
__global__ void cast_f32_f16_kernel(const float* __restrict__ in,
                                    _Float16* __restrict__ out, long n) {
    long i = (long)blockIdx.x * blockDim.x + threadIdx.x;
    if (i < n) out[i] = (_Float16)in[i];
}

// ---------------- LayerNorm (one token per 256-thread block) -> f16 ----------------
__global__ void ln_kernel(const float* __restrict__ x,
                          const float* __restrict__ gw,
                          const float* __restrict__ gb,
                          _Float16* __restrict__ out) {
    __shared__ float s1[256];
    __shared__ float s2[256];
    const int tid = threadIdx.x;
    const size_t t = blockIdx.x;
    const float* xp = x + t * D_;

    float v[4];
    float sum = 0.f, sq = 0.f;
#pragma unroll
    for (int j = 0; j < 4; ++j) {
        v[j] = xp[tid + j * 256];
        sum += v[j];
        sq  += v[j] * v[j];
    }
    s1[tid] = sum; s2[tid] = sq;
    __syncthreads();
    for (int s = 128; s > 0; s >>= 1) {
        if (tid < s) { s1[tid] += s1[tid + s]; s2[tid] += s2[tid + s]; }
        __syncthreads();
    }
    const float mu  = s1[0] * (1.f / D_);
    const float var = s2[0] * (1.f / D_) - mu * mu;
    const float rs  = rsqrtf(var + EPSf);
#pragma unroll
    for (int j = 0; j < 4; ++j) {
        const int i = tid + j * 256;
        out[t * D_ + i] = (_Float16)((v[j] - mu) * rs * gw[i] + gb[i]);
    }
}

// ---------------- WMMA fragment helpers ----------------
// A (16-bit 16x32): lane l holds row l&15; K runs [koff,koff+8) u [koff+16,koff+24),
// koff=(l>>4)*8  -> two 16B loads.
__device__ __forceinline__ v16h load_a_frag(const _Float16* __restrict__ p) {
    const v8h lo = *(const v8h*)(p);
    const v8h hi = *(const v8h*)(p + 16);
    v16h a;
#pragma unroll
    for (int i = 0; i < 8; ++i) { a[i] = lo[i]; a[i + 8] = hi[i]; }
    return a;
}

// ---------------- WMMA GEMM: C[M,N] = A[M,K(lda)] * W[N,K]^T ----------------
// One wave per 64x32 output tile (4x2 register tiling => 8 WMMAs per K-step,
// 6 fragments / 12 b128 loads per K-step => ~21 flop/byte of VMEM traffic).
// grid = (N/32, M/64), block = 32 (wave32).
// mode bit0: store f16 to Ch, else f32 to Cf.  mode bit1: softplus(acc + bias[n]).
__global__ void wmma_gemm_kernel(const _Float16* __restrict__ A, long lda,
                                 const _Float16* __restrict__ W, long ldw,
                                 float* __restrict__ Cf,
                                 _Float16* __restrict__ Ch, long ldc,
                                 const float* __restrict__ bias,
                                 long K, int mode) {
    const int lane = threadIdx.x;                 // 0..31
    const int m0 = blockIdx.y * (16 * MT);
    const int n0 = blockIdx.x * (16 * NT);
    const int half = lane >> 4;                   // 0 or 1
    const int l15  = lane & 15;

    const _Float16* ap[MT];
#pragma unroll
    for (int i = 0; i < MT; ++i)
        ap[i] = A + (size_t)(m0 + i * 16 + l15) * lda + (half << 3);
    // W (B operand): col = n_tile + (lane&15); same K split as A, koff=(l>>4)*8
    const _Float16* wp[NT];
#pragma unroll
    for (int j = 0; j < NT; ++j)
        wp[j] = W + (size_t)(n0 + j * 16 + l15) * ldw + (half << 3);

    v8f acc[MT][NT] = {};
    for (long k = 0; k < K; k += 32) {
        v16h a[MT], b[NT];
#pragma unroll
        for (int i = 0; i < MT; ++i) {
            __builtin_prefetch(ap[i] + k + 128, 0, 1);
            a[i] = load_a_frag(ap[i] + k);
        }
#pragma unroll
        for (int j = 0; j < NT; ++j) {
            __builtin_prefetch(wp[j] + k + 128, 0, 1);
            b[j] = load_a_frag(wp[j] + k);
        }
#pragma unroll
        for (int i = 0; i < MT; ++i)
#pragma unroll
            for (int j = 0; j < NT; ++j)
                acc[i][j] = __builtin_amdgcn_wmma_f32_16x16x32_f16(
                    false, a[i], false, b[j], (short)0, acc[i][j], false, false);
    }

    // C/D layout: VGPR r -> row (half*8 + r), col lane&15 within each 16x16 tile.
#pragma unroll
    for (int i = 0; i < MT; ++i) {
#pragma unroll
        for (int j = 0; j < NT; ++j) {
            const int n = n0 + (j << 4) + l15;
            const float bn = (mode & 2) ? bias[n] : 0.f;
#pragma unroll
            for (int r = 0; r < 8; ++r) {
                float v = acc[i][j][r];
                if (mode & 2) {                   // softplus(v + b)
                    v += bn;
                    v = (v > 20.f) ? v : log1pf(expf(v));
                }
                const size_t o = (size_t)(m0 + (i << 4) + (half << 3) + r) * ldc + n;
                if (mode & 1) Ch[o] = (_Float16)v;
                else          Cf[o] = v;
            }
        }
    }
}

// ---------------- depthwise causal conv (K=4) + SiLU, f16 in/out ----------------
__global__ void conv_silu_kernel(const _Float16* __restrict__ xz,   // [BT, 2*DI]
                                 const float* __restrict__ cw,      // [DI,1,4]
                                 const float* __restrict__ cb,      // [DI]
                                 _Float16* __restrict__ xout) {     // [BT, DI]
    const long total = (long)BT * DI;
    long i = (long)blockIdx.x * blockDim.x + threadIdx.x;
    if (i >= total) return;
    const int d = (int)(i % DI);
    const long bl = i / DI;
    const int l = (int)(bl % L_);
    const long brow = bl - l;                     // b*L
    float s = cb[d];
#pragma unroll
    for (int k = 0; k < Kcv; ++k) {
        const int ls = l - (Kcv - 1) + k;
        if (ls >= 0)
            s += cw[d * Kcv + k] * (float)xz[(brow + ls) * (size_t)(2 * DI) + d];
    }
    const float a = s / (1.f + __expf(-s));       // silu
    xout[bl * (size_t)DI + d] = (_Float16)a;
}

// ================= chunked parallel selective scan =================
// h_t = exp(dt_t*A) (.) h_{t-1} + (dt_t*x_t)*B_t  is associative in
// (decay, offset) pairs -> 3-pass chunk scan: 32x more parallelism than
// a serial L-walk (524288 threads instead of 16384 on passes A/C).

// Pass A: per-chunk local recurrence from h=0; emit final local state and
// the chunk decay product P. grid = (DI/256, NCH, B), block = 256.
__global__ void scan_chunk_local_kernel(const _Float16* __restrict__ dbl16, // [BT,96]
                                        const _Float16* __restrict__ dtf16, // [BT,DI]
                                        const _Float16* __restrict__ x16,   // [BT,DI]
                                        const float* __restrict__ A_log,    // [DI,16]
                                        float* __restrict__ P,              // [B,NCH,DI,16]
                                        float* __restrict__ Hloc) {         // [B,NCH,DI,16]
    __shared__ float sB[Nst];
    const int d = blockIdx.x * 256 + threadIdx.x;
    const int c = blockIdx.y;
    const int b = blockIdx.z;

    float Ad[Nst], h[Nst], p[Nst];
#pragma unroll
    for (int n = 0; n < Nst; ++n) {
        Ad[n] = -__expf(A_log[(size_t)d * Nst + n]);
        h[n] = 0.f; p[n] = 1.f;
    }

    for (int t = c * LC; t < (c + 1) * LC; ++t) {
        const size_t tok = (size_t)b * L_ + t;
        if (threadIdx.x < Nst)
            sB[threadIdx.x] = (float)dbl16[tok * 96 + DTR + threadIdx.x];
        __syncthreads();
        const float dt = (float)dtf16[tok * DI + d];
        const float dx = dt * (float)x16[tok * DI + d];
#pragma unroll
        for (int n = 0; n < Nst; ++n) {
            const float a = __expf(dt * Ad[n]);
            h[n] = a * h[n] + dx * sB[n];
            p[n] *= a;
        }
        __syncthreads();
    }
    const size_t o = ((size_t)(b * NCH + c) * DI + d) * Nst;
#pragma unroll
    for (int n = 0; n < Nst; ++n) { P[o + n] = p[n]; Hloc[o + n] = h[n]; }
}

// Pass B: inter-chunk prefix per (b,d,n); in-place converts Hloc[c] into the
// state ENTERING chunk c. grid covers B*DI threads.
__global__ void scan_chunk_prefix_kernel(const float* __restrict__ P,
                                         float* __restrict__ H) {  // Hloc -> Hinit
    const long i = (long)blockIdx.x * blockDim.x + threadIdx.x;   // over B*DI
    const int b = (int)(i / DI);
    const int d = (int)(i % DI);
    float acc[Nst];
#pragma unroll
    for (int n = 0; n < Nst; ++n) acc[n] = 0.f;
    for (int c = 0; c < NCH; ++c) {
        const size_t o = ((size_t)(b * NCH + c) * DI + d) * Nst;
#pragma unroll
        for (int n = 0; n < Nst; ++n) {
            const float hl = H[o + n];
            const float pc = P[o + n];
            H[o + n] = acc[n];               // state entering chunk c
            acc[n] = pc * acc[n] + hl;       // state entering chunk c+1
        }
    }
}

// Pass C: replay each chunk from its corrected initial state; emit
// y = (C_t . h_t) + x_t*D, gated by silu(z). grid = (DI/256, NCH, B).
__global__ void scan_chunk_emit_kernel(const _Float16* __restrict__ dbl16,
                                       const _Float16* __restrict__ dtf16,
                                       const _Float16* __restrict__ x16,
                                       const _Float16* __restrict__ xz16,  // z at +DI
                                       const float* __restrict__ A_log,
                                       const float* __restrict__ D_skip,
                                       const float* __restrict__ Hinit,    // [B,NCH,DI,16]
                                       _Float16* __restrict__ y16) {       // [BT,DI]
    __shared__ float sB[Nst];
    __shared__ float sC[Nst];
    const int d = blockIdx.x * 256 + threadIdx.x;
    const int c = blockIdx.y;
    const int b = blockIdx.z;

    const size_t o = ((size_t)(b * NCH + c) * DI + d) * Nst;
    float Ad[Nst], h[Nst];
#pragma unroll
    for (int n = 0; n < Nst; ++n) {
        Ad[n] = -__expf(A_log[(size_t)d * Nst + n]);
        h[n] = Hinit[o + n];
    }
    const float Dv = D_skip[d];

    for (int t = c * LC; t < (c + 1) * LC; ++t) {
        const size_t tok = (size_t)b * L_ + t;
        if (threadIdx.x < 2 * Nst) {
            const float v = (float)dbl16[tok * 96 + DTR + threadIdx.x];
            if (threadIdx.x < Nst) sB[threadIdx.x] = v;
            else                   sC[threadIdx.x - Nst] = v;
        }
        __syncthreads();
        const float dt = (float)dtf16[tok * DI + d];
        const float xt = (float)x16[tok * DI + d];
        const float dx = dt * xt;
        float y = 0.f;
#pragma unroll
        for (int n = 0; n < Nst; ++n) {
            h[n] = __expf(dt * Ad[n]) * h[n] + dx * sB[n];
            y += h[n] * sC[n];
        }
        y += xt * Dv;
        const float z = (float)xz16[tok * (size_t)(2 * DI) + DI + d];
        y *= z / (1.f + __expf(-z));
        y16[tok * DI + d] = (_Float16)y;
        __syncthreads();
    }
}

// ---------------- host side ----------------
extern "C" void kernel_launch(void* const* d_in, const int* in_sizes, int n_in,
                              void* d_out, int out_size, void* d_ws, size_t ws_size,
                              hipStream_t stream) {
    const float* hidden   = (const float*)d_in[0];   // [B,L,D]
    const float* norm_w   = (const float*)d_in[1];   // [D]
    const float* norm_b   = (const float*)d_in[2];   // [D]
    const float* in_w     = (const float*)d_in[3];   // [2*DI, D]
    const float* conv_w   = (const float*)d_in[4];   // [DI,1,4]
    const float* conv_b   = (const float*)d_in[5];   // [DI]
    const float* xproj_w  = (const float*)d_in[6];   // [96, 2048]
    const float* dtproj_w = (const float*)d_in[7];   // [DI, DTR]
    const float* dtproj_b = (const float*)d_in[8];   // [DI]
    const float* A_log    = (const float*)d_in[9];   // [DI, N]
    const float* D_skip   = (const float*)d_in[10];  // [DI]
    const float* out_w    = (const float*)d_in[11];  // [D, DI]
    float* out = (float*)d_out;                      // [BT,D] ++ [BT,D] residual

    // ---- workspace carve (256B aligned) ----
    size_t off = 0;
    auto carve = [&](size_t bytes) -> void* {
        void* p = (char*)d_ws + off;
        off += (bytes + 255) & ~(size_t)255;
        return p;
    };
    _Float16* h16    = (_Float16*)carve((size_t)BT * D_ * 2);        // 32 MB
    _Float16* w_in   = (_Float16*)carve((size_t)2 * DI * D_ * 2);    //  8 MB
    _Float16* xz16   = (_Float16*)carve((size_t)BT * 2 * DI * 2);    // 128 MB
    _Float16* x16    = (_Float16*)carve((size_t)BT * DI * 2);        // 64 MB
    _Float16* w_x    = (_Float16*)carve((size_t)96 * DI * 2);
    _Float16* dbl16  = (_Float16*)carve((size_t)BT * 96 * 2);        //  3 MB
    _Float16* w_dt   = (_Float16*)carve((size_t)DI * DTR * 2);
    _Float16* dtf16  = (_Float16*)carve((size_t)BT * DI * 2);        // 64 MB
    _Float16* y16    = (_Float16*)carve((size_t)BT * DI * 2);        // 64 MB
    _Float16* w_out  = (_Float16*)carve((size_t)D_ * DI * 2);        //  4 MB
    float*    scanP  = (float*)carve((size_t)B_ * NCH * DI * Nst * 4); // 32 MB
    float*    scanH  = (float*)carve((size_t)B_ * NCH * DI * Nst * 4); // 32 MB

    // ---- 1) weight down-conversion to f16 ----
    auto cast = [&](const float* src, _Float16* dst, long n) {
        cast_f32_f16_kernel<<<dim3((unsigned)((n + 255) / 256)), dim3(256), 0, stream>>>(src, dst, n);
    };
    cast(in_w,     w_in,  (long)2 * DI * D_);
    cast(xproj_w,  w_x,   (long)96 * DI);
    cast(dtproj_w, w_dt,  (long)DI * DTR);
    cast(out_w,    w_out, (long)D_ * DI);

    // ---- 2) LayerNorm -> h16 ----
    ln_kernel<<<dim3(BT), dim3(256), 0, stream>>>(hidden, norm_w, norm_b, h16);

    // ---- 3) in_proj: [BT,1024] x [4096,1024]^T -> xz16 [BT,4096] (f16) ----
    wmma_gemm_kernel<<<dim3(2 * DI / (16 * NT), BT / (16 * MT)), dim3(32), 0, stream>>>(
        h16, D_, w_in, D_, nullptr, xz16, 2 * DI, nullptr, D_, /*mode=*/1);

    // ---- 4) depthwise conv + silu -> x16 ----
    {
        const long total = (long)BT * DI;
        conv_silu_kernel<<<dim3((unsigned)((total + 255) / 256)), dim3(256), 0, stream>>>(
            xz16, conv_w, conv_b, x16);
    }

    // ---- 5) x_proj: [BT,2048] x [96,2048]^T -> dbl16 [BT,96] (f16) ----
    wmma_gemm_kernel<<<dim3(96 / (16 * NT), BT / (16 * MT)), dim3(32), 0, stream>>>(
        x16, DI, w_x, DI, nullptr, dbl16, 96, nullptr, DI, /*mode=*/1);

    // ---- 6) dt_proj: [BT,64] x [2048,64]^T, softplus(.+b) -> dtf16 [BT,2048] ----
    wmma_gemm_kernel<<<dim3(DI / (16 * NT), BT / (16 * MT)), dim3(32), 0, stream>>>(
        dbl16, 96, w_dt, DTR, nullptr, dtf16, DI, dtproj_b, DTR, /*mode=*/3);

    // ---- 7) chunked parallel scan + gate -> y16 ----
    scan_chunk_local_kernel<<<dim3(DI / 256, NCH, B_), dim3(256), 0, stream>>>(
        dbl16, dtf16, x16, A_log, scanP, scanH);
    scan_chunk_prefix_kernel<<<dim3((B_ * DI) / 256), dim3(256), 0, stream>>>(
        scanP, scanH);   // scanH becomes per-chunk initial states
    scan_chunk_emit_kernel<<<dim3(DI / 256, NCH, B_), dim3(256), 0, stream>>>(
        dbl16, dtf16, x16, xz16, A_log, D_skip, scanH, y16);

    // ---- 8) out_proj: [BT,2048] x [1024,2048]^T -> out (f32) ----
    wmma_gemm_kernel<<<dim3(D_ / (16 * NT), BT / (16 * MT)), dim3(32), 0, stream>>>(
        y16, DI, w_out, DI, out, nullptr, D_, nullptr, DI, /*mode=*/0);

    // ---- 9) residual = hidden_states (second tuple element) ----
    hipMemcpyAsync(out + (size_t)BT * D_, hidden, (size_t)BT * D_ * sizeof(float),
                   hipMemcpyDeviceToDevice, stream);
}
